// MFN_30030411333979
// MI455X (gfx1250) — matrix operand order repaired and checked by reading, not compiled, not run on hardware
//
#include <hip/hip_runtime.h>

// ---------------------------------------------------------------------------
// MFN (memory fusion network) for MI455X / gfx1250.
// bf16 WMMA (v_wmma_f32_16x16x32_bf16) for every GEMM, fp32 accumulate.
// Weights converted once to padded bf16 (~7MB -> resident in 192MB L2).
// GEMM is register-blocked: one wave owns a 16x64 output strip (4 WMMA
// accumulators) so each A fragment load is amortized over 4 WMMAs.
// ---------------------------------------------------------------------------

typedef __attribute__((ext_vector_type(16))) __bf16 v16bf;
typedef __attribute__((ext_vector_type(8)))  float  v8f;
typedef unsigned short u16;

__device__ __forceinline__ u16 f2bf(float x) {
    union { float f; unsigned u; } v; v.f = x;
    unsigned u = v.u;
    unsigned r = u + 0x7FFFu + ((u >> 16) & 1u);   // round-to-nearest-even
    return (u16)(r >> 16);
}
__device__ __forceinline__ float sigf(float x) { return 1.f / (1.f + expf(-x)); }

union FragBF { uint4 u[2]; v16bf v; };

// ---------------------------------------------------------------------------
// Dual-input WMMA GEMM:
//   C[512 x Nout] = act( A1[512 x K1]*W1[Nout x K1]^T
//                      + A2[512 x K2]*W2[Nout x K2]^T + bias )
// 128 threads = 4 waves. Wave w owns columns [(blockIdx.y*4+w)*64, +64):
// 4 accumulator tiles; A fragment loaded once per K step, 4 B fragments.
// blockIdx.x = 16-row tile (M=512 -> gridDim.x = 32).
// A-fragment in contiguous K order (lane half h holds K 16h..16h+15),
// B-fragment in the same K order (lane&15 = column, half selects K 0-15/16-31).
// ---------------------------------------------------------------------------
__global__ __launch_bounds__(128) void gemm_bf16_wmma(
    const u16* __restrict__ A1, int lda1,
    const u16* __restrict__ W1, int ldw1, int K1,
    const u16* __restrict__ A2, int lda2,
    const u16* __restrict__ W2, int ldw2, int K2,
    const float* __restrict__ bias, int act,
    float* __restrict__ Cf, u16* __restrict__ Cb, int Nout)
{
    const int lane    = threadIdx.x & 31;
    const int wave    = threadIdx.x >> 5;
    const int rowTile = blockIdx.x;
    const int half    = lane >> 4;
    const int l15     = lane & 15;
    const int row     = rowTile * 16 + l15;
    const int colBase = (blockIdx.y * 4 + wave) * 64;

    v8f acc0 = {}, acc1 = {}, acc2 = {}, acc3 = {};
    {
        const u16* aP  = A1 + (size_t)row * lda1 + half * 16;
        const u16* bP0 = W1 + (size_t)(colBase +  0 + l15) * ldw1 + half * 16;
        const u16* bP1 = W1 + (size_t)(colBase + 16 + l15) * ldw1 + half * 16;
        const u16* bP2 = W1 + (size_t)(colBase + 32 + l15) * ldw1 + half * 16;
        const u16* bP3 = W1 + (size_t)(colBase + 48 + l15) * ldw1 + half * 16;
        for (int k = 0; k < K1; k += 32) {
            __builtin_prefetch(aP + k + 128, 0, 1);
            __builtin_prefetch(bP0 + k + 128, 0, 1);
            FragBF a, b0, b1, b2, b3;
            a.u[0]  = *(const uint4*)(aP + k);
            a.u[1]  = *(const uint4*)(aP + k + 8);
            b0.u[0] = *(const uint4*)(bP0 + k);
            b0.u[1] = *(const uint4*)(bP0 + k + 8);
            b1.u[0] = *(const uint4*)(bP1 + k);
            b1.u[1] = *(const uint4*)(bP1 + k + 8);
            b2.u[0] = *(const uint4*)(bP2 + k);
            b2.u[1] = *(const uint4*)(bP2 + k + 8);
            b3.u[0] = *(const uint4*)(bP3 + k);
            b3.u[1] = *(const uint4*)(bP3 + k + 8);
            acc0 = __builtin_amdgcn_wmma_f32_16x16x32_bf16(false, a.v, false, b0.v, (short)0, acc0, false, false);
            acc1 = __builtin_amdgcn_wmma_f32_16x16x32_bf16(false, a.v, false, b1.v, (short)0, acc1, false, false);
            acc2 = __builtin_amdgcn_wmma_f32_16x16x32_bf16(false, a.v, false, b2.v, (short)0, acc2, false, false);
            acc3 = __builtin_amdgcn_wmma_f32_16x16x32_bf16(false, a.v, false, b3.v, (short)0, acc3, false, false);
        }
    }
    if (A2) {
        const u16* aP  = A2 + (size_t)row * lda2 + half * 16;
        const u16* bP0 = W2 + (size_t)(colBase +  0 + l15) * ldw2 + half * 16;
        const u16* bP1 = W2 + (size_t)(colBase + 16 + l15) * ldw2 + half * 16;
        const u16* bP2 = W2 + (size_t)(colBase + 32 + l15) * ldw2 + half * 16;
        const u16* bP3 = W2 + (size_t)(colBase + 48 + l15) * ldw2 + half * 16;
        for (int k = 0; k < K2; k += 32) {
            FragBF a, b0, b1, b2, b3;
            a.u[0]  = *(const uint4*)(aP + k);
            a.u[1]  = *(const uint4*)(aP + k + 8);
            b0.u[0] = *(const uint4*)(bP0 + k);
            b0.u[1] = *(const uint4*)(bP0 + k + 8);
            b1.u[0] = *(const uint4*)(bP1 + k);
            b1.u[1] = *(const uint4*)(bP1 + k + 8);
            b2.u[0] = *(const uint4*)(bP2 + k);
            b2.u[1] = *(const uint4*)(bP2 + k + 8);
            b3.u[0] = *(const uint4*)(bP3 + k);
            b3.u[1] = *(const uint4*)(bP3 + k + 8);
            acc0 = __builtin_amdgcn_wmma_f32_16x16x32_bf16(false, a.v, false, b0.v, (short)0, acc0, false, false);
            acc1 = __builtin_amdgcn_wmma_f32_16x16x32_bf16(false, a.v, false, b1.v, (short)0, acc1, false, false);
            acc2 = __builtin_amdgcn_wmma_f32_16x16x32_bf16(false, a.v, false, b2.v, (short)0, acc2, false, false);
            acc3 = __builtin_amdgcn_wmma_f32_16x16x32_bf16(false, a.v, false, b3.v, (short)0, acc3, false, false);
        }
    }
    v8f accs[4] = {acc0, acc1, acc2, acc3};
    #pragma unroll
    for (int i = 0; i < 4; ++i) {
        const int col = colBase + 16 * i + l15;
        const float bia = bias ? bias[col] : 0.f;
        #pragma unroll
        for (int v = 0; v < 8; ++v) {
            const int m = rowTile * 16 + half * 8 + v;  // D layout: lane half -> M+8
            float val = accs[i][v] + bia;
            if      (act == 1) val = fmaxf(val, 0.f);
            else if (act == 2) val = tanhf(val);
            else if (act == 3) val = sigf(val);
            if (Cf) Cf[(size_t)m * Nout + col] = val;
            if (Cb) Cb[(size_t)m * Nout + col] = f2bf(val);
        }
    }
}

// --------------------------- elementwise kernels ---------------------------

__global__ void pack_w(const float* __restrict__ src, u16* __restrict__ dst,
                       int R, int K0, int Kp) {
    int i = blockIdx.x * blockDim.x + threadIdx.x;
    if (i >= R * Kp) return;
    int r = i / Kp, k = i % Kp;
    dst[i] = f2bf(k < K0 ? src[(size_t)r * K0 + k] : 0.f);
}

// x step slice [512 x 410] -> padded bf16 [512 x 480]: t 0..319 (300), a 320..415 (74), v 416..479 (36)
__global__ void pack_x(const float* __restrict__ x, u16* __restrict__ xbf) {
    int i = blockIdx.x * blockDim.x + threadIdx.x;
    if (i >= 512 * 480) return;
    int n = i / 480, c = i % 480;
    float v = 0.f;
    if (c < 320)      { if (c < 300)              v = x[(size_t)n * 410 + c]; }
    else if (c < 416) { int s = c - 320; if (s < 74) v = x[(size_t)n * 410 + 300 + s]; }
    else              { int s = c - 416; if (s < 36) v = x[(size_t)n * 410 + 374 + s]; }
    xbf[i] = f2bf(v);
}

__global__ void zero_fbf(float* f, u16* b, int n) {
    int i = blockIdx.x * blockDim.x + threadIdx.x;
    if (i < n) { f[i] = 0.f; b[i] = 0; }
}
__global__ void cvt_f2bf(const float* s, u16* d, int n) {
    int i = blockIdx.x * blockDim.x + threadIdx.x;
    if (i < n) d[i] = f2bf(s[i]);
}

// Fused LSTM gate update for all three modalities (t:256, a:64, v:64).
// Also scatters old c (pre) and new c (cur) into c_star fp32+bf16.
__global__ void lstm_elem3(const float* __restrict__ z_t, const float* __restrict__ z_a,
                           const float* __restrict__ z_v,
                           float* __restrict__ tc, float* __restrict__ ac, float* __restrict__ vc,
                           float* __restrict__ th_f, float* __restrict__ ah_f, float* __restrict__ vh_f,
                           u16* __restrict__ th_bf, u16* __restrict__ ah_bf, u16* __restrict__ vh_bf,
                           float* __restrict__ cstar_f, u16* __restrict__ cstar_bf) {
    int idx = blockIdx.x * blockDim.x + threadIdx.x;
    const float* z; float* c; float* hf; u16* hb;
    int H, preOff, curOff;
    if (idx < 512 * 256) {
        z = z_t; c = tc; hf = th_f; hb = th_bf; H = 256; preOff = 0;   curOff = 384;
    } else if (idx < 512 * 320) {
        idx -= 512 * 256;
        z = z_a; c = ac; hf = ah_f; hb = ah_bf; H = 64;  preOff = 256; curOff = 640;
    } else if (idx < 512 * 384) {
        idx -= 512 * 320;
        z = z_v; c = vc; hf = vh_f; hb = vh_bf; H = 64;  preOff = 320; curOff = 704;
    } else return;
    int n = idx / H, j = idx % H;
    const float* zr = z + (size_t)n * 4 * H;
    float ig = sigf(zr[j]);
    float fg = sigf(zr[H + j]);
    float gg = tanhf(zr[2 * H + j]);
    float og = sigf(zr[3 * H + j]);
    float cold = c[idx];
    float c2 = fg * cold + ig * gg;
    float h2 = og * tanhf(c2);
    c[idx] = c2; hf[idx] = h2; hb[idx] = f2bf(h2);
    int p = n * 768 + preOff + j;
    int q = n * 768 + curOff + j;
    cstar_f[p] = cold; cstar_bf[p] = f2bf(cold);
    cstar_f[q] = c2;   cstar_bf[q] = f2bf(c2);
}

// row softmax over 768 logits, then attended = softmax * c_star (bf16 out)
__global__ void softmax_attend(const float* __restrict__ logits,
                               const float* __restrict__ cstar_f,
                               u16* __restrict__ attended_bf) {
    int row = blockIdx.x;
    const float* L = logits + (size_t)row * 768;
    __shared__ float red[256];
    int t = threadIdx.x;
    float mx = -1e30f;
    for (int j = t; j < 768; j += 256) mx = fmaxf(mx, L[j]);
    red[t] = mx; __syncthreads();
    for (int s = 128; s > 0; s >>= 1) { if (t < s) red[t] = fmaxf(red[t], red[t + s]); __syncthreads(); }
    mx = red[0]; __syncthreads();
    float sum = 0.f;
    for (int j = t; j < 768; j += 256) sum += expf(L[j] - mx);
    red[t] = sum; __syncthreads();
    for (int s = 128; s > 0; s >>= 1) { if (t < s) red[t] += red[t + s]; __syncthreads(); }
    float inv = 1.f / red[0];
    for (int j = t; j < 768; j += 256) {
        float p = expf(L[j] - mx) * inv;
        attended_bf[(size_t)row * 768 + j] = f2bf(p * cstar_f[(size_t)row * 768 + j]);
    }
}

// m = gamma1*m + gamma2*chat   (sigmoids & tanh fused in GEMM epilogues)
__global__ void mem_update(const float* __restrict__ g1o, const float* __restrict__ g2o,
                           const float* __restrict__ chat,
                           float* __restrict__ m_f, u16* __restrict__ m_bf) {
    int i = blockIdx.x * blockDim.x + threadIdx.x;
    if (i >= 512 * 256) return;
    float m = g1o[i] * m_f[i] + g2o[i] * chat[i];
    m_f[i] = m; m_bf[i] = f2bf(m);
}

// last_h = [th(256), ah(64), vh(64), mem(256)] -> bf16 [512 x 640]
__global__ void concat_lasth(const float* __restrict__ th, const float* __restrict__ ah,
                             const float* __restrict__ vh, const float* __restrict__ m,
                             u16* __restrict__ out) {
    int i = blockIdx.x * blockDim.x + threadIdx.x;
    if (i >= 512 * 640) return;
    int n = i / 640, c = i % 640;
    float v;
    if (c < 256)      v = th[(size_t)n * 256 + c];
    else if (c < 320) v = ah[(size_t)n * 64 + c - 256];
    else if (c < 384) v = vh[(size_t)n * 64 + c - 320];
    else              v = m[(size_t)n * 256 + c - 384];
    out[i] = f2bf(v);
}

// out[row] = dot(hid[row,:512], w2) + b2   (final 512->1 layer)
__global__ void final_dot(const float* __restrict__ hid, const float* __restrict__ w2,
                          const float* __restrict__ b2, float* __restrict__ out) {
    int row = blockIdx.x, t = threadIdx.x;
    __shared__ float red[256];
    float s = 0.f;
    for (int j = t; j < 512; j += 256) s += hid[(size_t)row * 512 + j] * w2[j];
    red[t] = s; __syncthreads();
    for (int k = 128; k > 0; k >>= 1) { if (t < k) red[t] += red[t + k]; __syncthreads(); }
    if (t == 0) out[row] = red[0] + b2[0];
}

// ---------------------------------------------------------------------------

extern "C" void kernel_launch(void* const* d_in, const int* in_sizes, int n_in,
                              void* d_out, int out_size, void* d_ws, size_t ws_size,
                              hipStream_t stream) {
    (void)in_sizes; (void)n_in; (void)out_size; (void)ws_size;
    const float* x_p   = (const float*)d_in[0];
    const float* c_t0  = (const float*)d_in[1];
    const float* c_a0  = (const float*)d_in[2];
    const float* c_v0  = (const float*)d_in[3];
    const float* mem0  = (const float*)d_in[4];
    const float* t_Wi  = (const float*)d_in[5];  const float* t_Wh = (const float*)d_in[6];  const float* t_b = (const float*)d_in[7];
    const float* a_Wi  = (const float*)d_in[8];  const float* a_Wh = (const float*)d_in[9];  const float* a_b = (const float*)d_in[10];
    const float* v_Wi  = (const float*)d_in[11]; const float* v_Wh = (const float*)d_in[12]; const float* v_b = (const float*)d_in[13];
    const float* a1w1 = (const float*)d_in[14]; const float* a1b1 = (const float*)d_in[15];
    const float* a1w2 = (const float*)d_in[16]; const float* a1b2 = (const float*)d_in[17];
    const float* a2w1 = (const float*)d_in[18]; const float* a2b1 = (const float*)d_in[19];
    const float* a2w2 = (const float*)d_in[20]; const float* a2b2 = (const float*)d_in[21];
    const float* g1w1 = (const float*)d_in[22]; const float* g1b1 = (const float*)d_in[23];
    const float* g1w2 = (const float*)d_in[24]; const float* g1b2 = (const float*)d_in[25];
    const float* g2w1 = (const float*)d_in[26]; const float* g2b1 = (const float*)d_in[27];
    const float* g2w2 = (const float*)d_in[28]; const float* g2b2 = (const float*)d_in[29];
    const float* ow1  = (const float*)d_in[30]; const float* ob1  = (const float*)d_in[31];
    const float* ow2  = (const float*)d_in[32]; const float* ob2  = (const float*)d_in[33];
    float* out = (float*)d_out;

    // ---- workspace carve-up ----
    char* ws = (char*)d_ws;
    size_t off = 0;
    auto alloc = [&](size_t bytes) -> void* {
        off = (off + 255) & ~(size_t)255;
        void* p = ws + off;
        off += bytes;
        return p;
    };
    // bf16 weights (padded)
    u16* tWi_bf  = (u16*)alloc(1024u * 320 * 2);
    u16* tWh_bf  = (u16*)alloc(1024u * 256 * 2);
    u16* aWi_bf  = (u16*)alloc(256u * 96 * 2);
    u16* aWh_bf  = (u16*)alloc(256u * 64 * 2);
    u16* vWi_bf  = (u16*)alloc(256u * 64 * 2);
    u16* vWh_bf  = (u16*)alloc(256u * 64 * 2);
    u16* a1w1_bf = (u16*)alloc(512u * 768 * 2);
    u16* a1w2_bf = (u16*)alloc(768u * 512 * 2);
    u16* a2w1_bf = (u16*)alloc(512u * 768 * 2);
    u16* a2w2_bf = (u16*)alloc(256u * 512 * 2);
    u16* g1w1_bf = (u16*)alloc(512u * 1024 * 2);
    u16* g1w2_bf = (u16*)alloc(256u * 512 * 2);
    u16* g2w1_bf = (u16*)alloc(512u * 1024 * 2);
    u16* g2w2_bf = (u16*)alloc(256u * 512 * 2);
    u16* ow1_bf  = (u16*)alloc(512u * 640 * 2);
    // fp32 state / activations
    float* tc    = (float*)alloc(512u * 256 * 4);
    float* ac    = (float*)alloc(512u * 64 * 4);
    float* vc    = (float*)alloc(512u * 64 * 4);
    float* th_f  = (float*)alloc(512u * 256 * 4);
    float* ah_f  = (float*)alloc(512u * 64 * 4);
    float* vh_f  = (float*)alloc(512u * 64 * 4);
    float* mem_f = (float*)alloc(512u * 256 * 4);
    float* z_t   = (float*)alloc(512u * 1024 * 4);
    float* z_a   = (float*)alloc(512u * 256 * 4);
    float* z_v   = (float*)alloc(512u * 256 * 4);
    float* cstar_f = (float*)alloc(512u * 768 * 4);
    float* logits  = (float*)alloc(512u * 768 * 4);
    float* chat    = (float*)alloc(512u * 256 * 4);
    float* g1o     = (float*)alloc(512u * 256 * 4);
    float* g2o     = (float*)alloc(512u * 256 * 4);
    float* hid_f   = (float*)alloc(512u * 512 * 4);
    // bf16 activations
    u16* xbf      = (u16*)alloc(512u * 480 * 2);
    u16* th_bf    = (u16*)alloc(512u * 256 * 2);
    u16* ah_bf    = (u16*)alloc(512u * 64 * 2);
    u16* vh_bf    = (u16*)alloc(512u * 64 * 2);
    u16* mem_bf   = (u16*)alloc(512u * 256 * 2);
    u16* cstar_bf = (u16*)alloc(512u * 768 * 2);
    u16* att_bf   = (u16*)alloc(512u * 768 * 2);
    u16* h1_bf    = (u16*)alloc(512u * 512 * 2);
    u16* lasth_bf = (u16*)alloc(512u * 640 * 2);

    auto packW = [&](const float* src, u16* dst, int R, int K0, int Kp) {
        int n = R * Kp;
        pack_w<<<(n + 255) / 256, 256, 0, stream>>>(src, dst, R, K0, Kp);
    };
    auto gemm = [&](const u16* A1, int lda1, const u16* W1, int ldw1, int K1,
                    const u16* A2, int lda2, const u16* W2, int ldw2, int K2,
                    const float* bias, int act, float* Cf, u16* Cb, int Nout) {
        dim3 grid(32, Nout / 256);   // one wave = 16x64 strip; 4 waves = 256 cols
        gemm_bf16_wmma<<<grid, 128, 0, stream>>>(A1, lda1, W1, ldw1, K1,
                                                 A2, lda2, W2, ldw2, K2,
                                                 bias, act, Cf, Cb, Nout);
    };

    // ---- one-time per call: weight conversion + state init ----
    packW(t_Wi, tWi_bf, 1024, 300, 320);
    packW(t_Wh, tWh_bf, 1024, 256, 256);
    packW(a_Wi, aWi_bf, 256, 74, 96);
    packW(a_Wh, aWh_bf, 256, 64, 64);
    packW(v_Wi, vWi_bf, 256, 36, 64);
    packW(v_Wh, vWh_bf, 256, 64, 64);
    packW(a1w1, a1w1_bf, 512, 768, 768);
    packW(a1w2, a1w2_bf, 768, 512, 512);
    packW(a2w1, a2w1_bf, 512, 768, 768);
    packW(a2w2, a2w2_bf, 256, 512, 512);
    packW(g1w1, g1w1_bf, 512, 1024, 1024);
    packW(g1w2, g1w2_bf, 256, 512, 512);
    packW(g2w1, g2w1_bf, 512, 1024, 1024);
    packW(g2w2, g2w2_bf, 256, 512, 512);
    packW(ow1,  ow1_bf,  512, 640, 640);

    hipMemcpyAsync(tc, c_t0, 512u * 256 * 4, hipMemcpyDeviceToDevice, stream);
    hipMemcpyAsync(ac, c_a0, 512u * 64 * 4, hipMemcpyDeviceToDevice, stream);
    hipMemcpyAsync(vc, c_v0, 512u * 64 * 4, hipMemcpyDeviceToDevice, stream);
    hipMemcpyAsync(mem_f, mem0, 512u * 256 * 4, hipMemcpyDeviceToDevice, stream);
    zero_fbf<<<(512 * 256 + 255) / 256, 256, 0, stream>>>(th_f, th_bf, 512 * 256);
    zero_fbf<<<(512 * 64 + 255) / 256, 256, 0, stream>>>(ah_f, ah_bf, 512 * 64);
    zero_fbf<<<(512 * 64 + 255) / 256, 256, 0, stream>>>(vh_f, vh_bf, 512 * 64);
    cvt_f2bf<<<(512 * 256 + 255) / 256, 256, 0, stream>>>(mem_f, mem_bf, 512 * 256);

    // ---- 128 sequential timesteps ----
    for (int t = 0; t < 128; ++t) {
        const float* xt = x_p + (size_t)t * 512 * 410;
        pack_x<<<(512 * 480 + 255) / 256, 256, 0, stream>>>(xt, xbf);

        // LSTM pre-activations: z = x*Wi^T + h*Wh^T + b
        gemm(xbf, 480, tWi_bf, 320, 320, th_bf, 256, tWh_bf, 256, 256, t_b, 0, z_t, nullptr, 1024);
        gemm(xbf + 320, 480, aWi_bf, 96, 96, ah_bf, 64, aWh_bf, 64, 64, a_b, 0, z_a, nullptr, 256);
        gemm(xbf + 416, 480, vWi_bf, 64, 64, vh_bf, 64, vWh_bf, 64, 64, v_b, 0, z_v, nullptr, 256);

        // fused gates + c_star = [pre_c(384), cur_c(384)]
        lstm_elem3<<<(512 * 384 + 255) / 256, 256, 0, stream>>>(
            z_t, z_a, z_v, tc, ac, vc, th_f, ah_f, vh_f, th_bf, ah_bf, vh_bf,
            cstar_f, cstar_bf);

        // attention MLP1 (768 -> 512 relu -> 768), softmax, attend
        gemm(cstar_bf, 768, a1w1_bf, 768, 768, nullptr, 0, nullptr, 0, 0, a1b1, 1, nullptr, h1_bf, 512);
        gemm(h1_bf, 512, a1w2_bf, 512, 512, nullptr, 0, nullptr, 0, 0, a1b2, 0, logits, nullptr, 768);
        softmax_attend<<<512, 256, 0, stream>>>(logits, cstar_f, att_bf);

        // c_hat = tanh(MLP2(attended)) : 768 -> 512 relu -> 256 tanh
        gemm(att_bf, 768, a2w1_bf, 768, 768, nullptr, 0, nullptr, 0, 0, a2b1, 1, nullptr, h1_bf, 512);
        gemm(h1_bf, 512, a2w2_bf, 512, 512, nullptr, 0, nullptr, 0, 0, a2b2, 2, chat, nullptr, 256);

        // gamma1 = sigmoid(MLP(both)), both = [attended(768), mem(256)] via dual-A
        gemm(att_bf, 768, g1w1_bf, 1024, 768, mem_bf, 256, g1w1_bf + 768, 1024, 256, g1b1, 1, nullptr, h1_bf, 512);
        gemm(h1_bf, 512, g1w2_bf, 512, 512, nullptr, 0, nullptr, 0, 0, g1b2, 3, g1o, nullptr, 256);
        // gamma2
        gemm(att_bf, 768, g2w1_bf, 1024, 768, mem_bf, 256, g2w1_bf + 768, 1024, 256, g2b1, 1, nullptr, h1_bf, 512);
        gemm(h1_bf, 512, g2w2_bf, 512, 512, nullptr, 0, nullptr, 0, 0, g2b2, 3, g2o, nullptr, 256);

        mem_update<<<(512 * 256 + 255) / 256, 256, 0, stream>>>(g1o, g2o, chat, mem_f, mem_bf);
    }

    // ---- output head ----
    concat_lasth<<<(512 * 640 + 255) / 256, 256, 0, stream>>>(th_f, ah_f, vh_f, mem_f, lasth_bf);
    gemm(lasth_bf, 640, ow1_bf, 640, 640, nullptr, 0, nullptr, 0, 0, ob1, 1, hid_f, nullptr, 512);
    final_dot<<<512, 256, 0, stream>>>(hid_f, ow2, ob2, out);
}
